// SpectralMamba_14388140441602
// MI455X (gfx1250) — compile-verified, matrix operand order checked by reading, not run on hardware
//
#include <hip/hip_runtime.h>
#include <hip/hip_bf16.h>
#include <cstddef>

// ---------------------------------------------------------------------------
// Types
// ---------------------------------------------------------------------------
typedef __bf16 bf16_t;
typedef __attribute__((ext_vector_type(16))) bf16_t v16bf;
typedef __attribute__((ext_vector_type(8)))  bf16_t v8bf;
typedef __attribute__((ext_vector_type(8)))  float  v8f;

#define D_MODEL 256
#define D_INNER 512
#define HW_TOT  16384   // 128*128
#define TOK     16      // tokens per workgroup (WMMA N dim)

// Packed-weight element counts (bf16 elements) in d_ws
#define N_W1   (1024 * 256)   // W_in^T  : M=1024, Kpad=256
#define N_WXP  (48 * 512)     // W_xproj^T: M=48,  Kpad=512
#define N_WDT  (512 * 32)     // W_dt^T  : M=512, Kpad=32 (Korig=16, zero padded)
#define N_WOUT (256 * 512)    // W_out^T : M=256, Kpad=512
#define N_PACK_TOTAL (N_W1 + N_WXP + N_WDT + N_WOUT)

// ---------------------------------------------------------------------------
// Weight prepack: store W^T tiles in the exact 16-bit A-matrix VGPR layout of
// V_WMMA_*_16X16X32 (ISA 7.12.2): lane -> M row, element e -> K with
//   k = (e>>3)*16 + (lane>>4)*8 + (e&7)
// Packed index = ((mt*Kt + kt)*32 + lane)*16 + e  (bf16 elements), so the main
// kernel fetches one A fragment as a single aligned 32-byte vector load.
// All source weights are stored [K][M] row-major (K = input dim).
// ---------------------------------------------------------------------------
__global__ void prepack_kernel(const float* __restrict__ Win,
                               const float* __restrict__ Wxp,
                               const float* __restrict__ Wdt,
                               const float* __restrict__ Wout,
                               bf16_t* __restrict__ ws)
{
    int idx = blockIdx.x * blockDim.x + threadIdx.x;
    if (idx >= N_PACK_TOTAL) return;

    const float* src;
    bf16_t* dst;
    int M, Kt, Korig, li;

    if (idx < N_W1) {
        src = Win;  dst = ws;                        M = 1024; Kt = 8;  Korig = 256; li = idx;
    } else if (idx < N_W1 + N_WXP) {
        src = Wxp;  dst = ws + N_W1;                 M = 48;   Kt = 16; Korig = 512; li = idx - N_W1;
    } else if (idx < N_W1 + N_WXP + N_WDT) {
        src = Wdt;  dst = ws + N_W1 + N_WXP;         M = 512;  Kt = 1;  Korig = 16;  li = idx - N_W1 - N_WXP;
    } else {
        src = Wout; dst = ws + N_W1 + N_WXP + N_WDT; M = 256;  Kt = 16; Korig = 512; li = idx - N_W1 - N_WXP - N_WDT;
    }

    int tile = li >> 9;          // 512 elements per tile (32 lanes * 16)
    int r    = li & 511;
    int lane = r >> 4;
    int e    = r & 15;
    int mt   = tile / Kt;
    int kt   = tile - mt * Kt;

    int m  = mt * 16 + (lane & 15);
    int hf = lane >> 4;
    int k  = kt * 32 + ((e >> 3) << 4) + (hf << 3) + (e & 7);

    float v = (k < Korig) ? src[(size_t)k * M + m] : 0.f;
    dst[li] = (bf16_t)v;
}

// ---------------------------------------------------------------------------
// B fragment (K x 16 activations) gather from an LDS row act[n][0..K):
// per lane the 16 elements are two contiguous runs of 8 -> two ds_load_b128.
// Row strides (K+8 bf16) keep the 16 lanes on distinct bank groups.
// ---------------------------------------------------------------------------
__device__ __forceinline__ v16bf load_bfrag(const bf16_t* row, int kt, int hf)
{
    int k0 = kt * 32 + hf * 8;
    v8bf lo = *(const v8bf*)(row + k0);
    v8bf hi = *(const v8bf*)(row + k0 + 16);
    return __builtin_shufflevector(lo, hi, 0, 1, 2, 3, 4, 5, 6, 7,
                                           8, 9, 10, 11, 12, 13, 14, 15);
}

// Fast silu: one v_exp + one v_rcp (no IEEE divide expansion).
__device__ __forceinline__ float silu_f(float t)
{
    return t * __builtin_amdgcn_rcpf(1.f + __expf(-t));
}

// ---------------------------------------------------------------------------
// Fused Mamba-L1 block: one workgroup = 16 tokens (b fixed, hw contiguous).
// 8 wave32s.  All GEMMs via v_wmma_f32_16x16x32_bf16, f32 accumulate.
// ---------------------------------------------------------------------------
__global__ void __launch_bounds__(256)
mamba_main(const float* __restrict__ x,
           const float* __restrict__ conv_w,   // [512,4], use column 3
           const float* __restrict__ conv_b,   // [512]
           const float* __restrict__ b_dt,     // [512]
           const float* __restrict__ Dp,       // [512]
           const bf16_t* __restrict__ wsb,     // packed weights
           float* __restrict__ out)
{
    __shared__ __align__(16) bf16_t u_lds[TOK][264];   // u tile, K=256 (+8 pad)
    __shared__ __align__(16) bf16_t a_lds[TOK][520];   // xc, later y (in place)
    __shared__ __align__(16) bf16_t z_lds[TOK][520];   // silu(z)
    __shared__ __align__(16) bf16_t dt_lds[TOK][40];   // dt, K padded to 32
    __shared__ float bm_lds[16][TOK];
    __shared__ float cm_lds[16][TOK];
    __shared__ float bc_lds[TOK];

    const v16bf* w1f  = (const v16bf*)(wsb);
    const v16bf* wxpf = (const v16bf*)(wsb + N_W1);
    const v16bf* wdtf = (const v16bf*)(wsb + N_W1 + N_WXP);
    const v16bf* wof  = (const v16bf*)(wsb + N_W1 + N_WXP + N_WDT);

    const int tid  = threadIdx.x;
    const int w    = tid >> 5;        // wave id 0..7
    const int lane = tid & 31;
    const int hf   = lane >> 4;       // half-wave select
    const int ln   = lane & 15;       // token (N) index within tile

    const int tile = blockIdx.x;                 // 8192 tiles
    const int bb   = tile >> 10;                 // batch (1024 tiles per batch)
    const int hw0  = (tile & 1023) << 4;         // 16 consecutive hw positions
    const float* xb = x + (size_t)bb * D_MODEL * HW_TOT + hw0;

    // ---- Stage 0: load u tile (f32 -> bf16), coalesced 64B per channel row
    {
        int n  = tid & 15;
        int c0 = tid >> 4;
        #pragma unroll
        for (int i = 0; i < 16; ++i) {
            int c = c0 + (i << 4);
            u_lds[n][c] = (bf16_t)xb[(size_t)c * HW_TOT + n];
        }
        // zero the K-padding of dt (rank 16 padded to K=32)
        int nn = tid >> 4, j = tid & 15;
        dt_lds[nn][16 + j] = (bf16_t)0.f;
        if (j < 8) dt_lds[nn][32 + j] = (bf16_t)0.f;
    }
    __syncthreads();

    // ---- Stage 1: xz^T[1024x16] = W_in^T @ u^T ; fuse conv-scale + silu
    {
        v16bf bu[8];
        #pragma unroll
        for (int kt = 0; kt < 8; ++kt) bu[kt] = load_bfrag(&u_lds[ln][0], kt, hf);

        #pragma unroll
        for (int j = 0; j < 8; ++j) {
            int mt = w * 8 + j;
            v8f acc = {};
            #pragma unroll
            for (int kt = 0; kt < 8; ++kt) {
                v16bf a = w1f[(mt * 8 + kt) * 32 + lane];
                acc = __builtin_amdgcn_wmma_f32_16x16x32_bf16(
                    false, a, false, bu[kt], (short)0, acc, false, false);
            }
            int f0 = mt * 16;
            if (f0 < D_INNER) {                       // xi -> xc  (waves 0..3)
                #pragma unroll
                for (int v = 0; v < 8; ++v) {
                    int fm  = f0 + hf * 8 + v;
                    float t = acc[v] * conv_w[fm * 4 + 3] + conv_b[fm];
                    a_lds[ln][fm] = (bf16_t)silu_f(t);
                }
            } else {                                  // z -> silu(z) (waves 4..7)
                #pragma unroll
                for (int v = 0; v < 8; ++v) {
                    int fm = f0 + hf * 8 + v;
                    z_lds[ln][fm - D_INNER] = (bf16_t)silu_f(acc[v]);
                }
            }
        }
    }
    __syncthreads();

    // ---- Stage 2: dbc^T[48x16] = W_xproj^T @ xc^T (waves 0..2)
    if (w < 3) {
        v8f acc = {};
        #pragma unroll
        for (int kt = 0; kt < 16; ++kt) {
            v16bf bfrag = load_bfrag(&a_lds[ln][0], kt, hf);
            v16bf a     = wxpf[(w * 16 + kt) * 32 + lane];
            acc = __builtin_amdgcn_wmma_f32_16x16x32_bf16(
                false, a, false, bfrag, (short)0, acc, false, false);
        }
        #pragma unroll
        for (int v = 0; v < 8; ++v) {
            int s = hf * 8 + v;
            if (w == 0)      dt_lds[ln][s] = (bf16_t)acc[v]; // dt
            else if (w == 1) bm_lds[s][ln] = acc[v];         // B
            else             cm_lds[s][ln] = acc[v];         // C
        }
    }
    __syncthreads();

    // ---- bc[n] = sum_s B[n][s]*C[n][s]  (16 threads), overlapped with stage 3
    if (tid < 16) {
        float s = 0.f;
        #pragma unroll
        for (int i = 0; i < 16; ++i) s += bm_lds[i][tid] * cm_lds[i][tid];
        bc_lds[tid] = s;
    }

    // ---- Stage 3: delta_pre^T[512x16] = W_dt^T @ dt^T (K padded to 32)
    v16bf bd = load_bfrag(&dt_lds[ln][0], 0, hf);
    v8f acc3[4];
    #pragma unroll
    for (int j = 0; j < 4; ++j) {
        int mt = w * 4 + j;
        v8f zc = {};
        acc3[j] = __builtin_amdgcn_wmma_f32_16x16x32_bf16(
            false, wdtf[mt * 32 + lane], false, bd, (short)0, zc, false, false);
    }
    __syncthreads();

    // ---- Fused elementwise: y = (softplus(.)*bc + D)*xc*silu(z), in-place
    {
        float bcv = bc_lds[ln];
        #pragma unroll
        for (int j = 0; j < 4; ++j) {
            int f0 = (w * 4 + j) * 16;
            #pragma unroll
            for (int v = 0; v < 8; ++v) {
                int fm   = f0 + hf * 8 + v;
                float t  = acc3[j][v] + b_dt[fm];
                float dl = (t > 15.f) ? t : __logf(1.f + __expf(t));  // softplus
                float xc = (float)a_lds[ln][fm];
                float yv = (dl * bcv + Dp[fm]) * xc * (float)z_lds[ln][fm];
                a_lds[ln][fm] = (bf16_t)yv;   // y overwrites xc (same owner)
            }
        }
    }
    __syncthreads();

    // ---- Stage 4: out^T[256x16] = W_out^T @ y^T ; kt-outer with two live
    //      accumulators so each B fragment is loaded from LDS exactly once.
    {
        float* ob = out + (size_t)bb * D_MODEL * HW_TOT + hw0;
        int mt0 = w * 2;
        v8f acc0 = {}, acc1 = {};
        #pragma unroll
        for (int kt = 0; kt < 16; ++kt) {
            v16bf bfrag = load_bfrag(&a_lds[ln][0], kt, hf);
            v16bf a0    = wof[(mt0 * 16 + kt) * 32 + lane];
            v16bf a1    = wof[((mt0 + 1) * 16 + kt) * 32 + lane];
            acc0 = __builtin_amdgcn_wmma_f32_16x16x32_bf16(
                false, a0, false, bfrag, (short)0, acc0, false, false);
            acc1 = __builtin_amdgcn_wmma_f32_16x16x32_bf16(
                false, a1, false, bfrag, (short)0, acc1, false, false);
        }
        #pragma unroll
        for (int v = 0; v < 8; ++v) {
            int fm0 = mt0 * 16 + hf * 8 + v;
            ob[(size_t)fm0 * HW_TOT + ln]            = acc0[v];
            ob[(size_t)(fm0 + 16) * HW_TOT + ln]     = acc1[v];
        }
    }
}

// ---------------------------------------------------------------------------
// Launch: prepack weights into d_ws (needs 868352 bytes), then fused kernel.
// ---------------------------------------------------------------------------
extern "C" void kernel_launch(void* const* d_in, const int* in_sizes, int n_in,
                              void* d_out, int out_size, void* d_ws, size_t ws_size,
                              hipStream_t stream)
{
    const float* x      = (const float*)d_in[0];
    const float* W_in   = (const float*)d_in[1];
    const float* conv_w = (const float*)d_in[2];
    const float* conv_b = (const float*)d_in[3];
    const float* W_xp   = (const float*)d_in[4];
    const float* W_dt   = (const float*)d_in[5];
    const float* b_dt   = (const float*)d_in[6];
    // d_in[7] = A_log: unused by the reference math
    const float* Dp     = (const float*)d_in[8];
    const float* W_out  = (const float*)d_in[9];
    (void)in_sizes; (void)n_in; (void)out_size; (void)ws_size;

    bf16_t* wsb = (bf16_t*)d_ws;

    prepack_kernel<<<(N_PACK_TOTAL + 255) / 256, 256, 0, stream>>>(
        W_in, W_xp, W_dt, W_out, wsb);

    int ntiles = 8 * (HW_TOT / TOK);   // 8192 workgroups, 16 tokens each
    mamba_main<<<ntiles, 256, 0, stream>>>(
        x, conv_w, conv_b, b_dt, Dp, wsb, (float*)d_out);
}
